// Decoder_71700184039955
// MI455X (gfx1250) — matrix operand order repaired and checked by reading, not compiled, not run on hardware
//
#include <hip/hip_runtime.h>
#include <hip/hip_bf16.h>
#include <math.h>

typedef __bf16 bf16;
typedef __attribute__((ext_vector_type(4)))  __bf16 v4bf;
typedef __attribute__((ext_vector_type(8)))  __bf16 v8bf;
typedef __attribute__((ext_vector_type(16))) __bf16 v16bf;
typedef __attribute__((ext_vector_type(8)))  float  v8f;

static constexpr int Bn = 64;    // batch
static constexpr int Tn = 256;   // target len
static constexpr int Sn = 512;   // source len
static constexpr int En = 128;   // embed
static constexpr int Hn = 512;   // hidden
static constexpr int H2 = 1024;  // 2H
static constexpr int G4 = 2048;  // 4H (gates)
static constexpr int KX = En + H2;        // 1152  (LSTM input K)
static constexpr int KP = En + Hn + H2;   // 1664  (pre K)

// ---------------- wave helpers (wave32) ----------------
__device__ __forceinline__ float wave_sum(float v) {
#pragma unroll
  for (int o = 16; o; o >>= 1) v += __shfl_xor(v, o, 32);
  return v;
}
__device__ __forceinline__ float wave_max(float v) {
#pragma unroll
  for (int o = 16; o; o >>= 1) v = fmaxf(v, __shfl_xor(v, o, 32));
  return v;
}
__device__ __forceinline__ float sigm(float x) { return 1.0f / (1.0f + __expf(-x)); }

// ---------------- WMMA bf16 fragment loaders (16x16x32, contiguous k order shared by A and B) ---------
// A 16x32 (MxK): lanes 0-15 row M=lane hold K = k0+0..15 contiguous;
// lanes 16-31 hold K = k0+16..31.
__device__ __forceinline__ v16bf load_a_frag(const bf16* __restrict__ A, int lda,
                                             int m0, int k0, int lane) {
  const int row = m0 + (lane & 15);
  const int kb  = k0 + ((lane & 16) ? 16 : 0);
  const bf16* p = A + (size_t)row * lda + kb;
  v8bf lo = *reinterpret_cast<const v8bf*>(p);
  v8bf hi = *reinterpret_cast<const v8bf*>(p + 8);
  return __builtin_shufflevector(lo, hi, 0, 1, 2, 3, 4, 5, 6, 7,
                                 8, 9, 10, 11, 12, 13, 14, 15);
}
// B 32x16 (KxN): lane<16 holds column N=n0+lane, K = k0+0..15 contiguous;
// lane>=16 holds column N=n0+lane-16, K = k0+16..31. Weights are (N,K)
// row-major, so each lane reads 16 contiguous bf16 (32 bytes).
__device__ __forceinline__ v16bf load_b_frag(const bf16* __restrict__ W, int ldb,
                                             int n0, int k0, int lane) {
  const int col = n0 + (lane & 15);
  const int kb  = k0 + ((lane & 16) ? 16 : 0);
  return *reinterpret_cast<const v16bf*>(W + (size_t)col * ldb + kb);
}
__device__ __forceinline__ v8f wmma_bf16(v16bf a, v16bf b, v8f c) {
  return __builtin_amdgcn_wmma_f32_16x16x32_bf16(false, a, false, b,
                                                 (short)0, c, false, false);
}

// ---------------- async global->LDS copy (CDNA5, ASYNCcnt-tracked) -----------
__device__ __forceinline__ void async_b128(unsigned lds_byte, const void* gaddr) {
  asm volatile("global_load_async_to_lds_b128 %0, %1, off"
               :: "v"(lds_byte), "v"(gaddr) : "memory");
}
#define WAIT_ASYNC_LE(n) asm volatile("s_wait_asynccnt " #n ::: "memory")

// ---------------- big NT GEMM: C(M,N) = A(M,K) * W(N,K)^T --------------------
// wave-per-(16x64) tile; async double-buffered LDS staging feeding WMMA.
// LDS/block = 2*(1KB A + 4KB B) = 10KB -> up to 32 blocks/WGP.
template <typename CT>
__global__ __launch_bounds__(32) void gemm_bf16_async(
    const bf16* __restrict__ A, int lda, const bf16* __restrict__ W, int ldb,
    CT* __restrict__ C, int ldc, int K) {
  __shared__ __align__(32) bf16 shA[2][16 * 32];
  __shared__ __align__(32) bf16 shB[2][64 * 32];
  const int lane = threadIdx.x;
  const int m0 = blockIdx.x * 16;
  const int n0 = blockIdx.y * 64;
  const unsigned ldsA0 = (unsigned)(size_t)(&shA[0][0]);
  const unsigned ldsB0 = (unsigned)(size_t)(&shB[0][0]);

  auto stage = [&](int buf, int k0) {
    // A tile: 16 rows x 64B  (2 async b128 per lane)
#pragma unroll
    for (int i = 0; i < 2; ++i) {
      int c = lane + 32 * i;                    // chunk 0..63
      int row = c >> 2, q = c & 3;
      const char* g = (const char*)(A + (size_t)(m0 + row) * lda + k0) + q * 16;
      async_b128(ldsA0 + (unsigned)(buf * 1024 + c * 16), g);
    }
    // B tile: 64 rows x 64B  (8 async b128 per lane)
#pragma unroll
    for (int i = 0; i < 8; ++i) {
      int c = lane + 32 * i;                    // chunk 0..255
      int row = c >> 2, q = c & 3;
      const char* g = (const char*)(W + (size_t)(n0 + row) * ldb + k0) + q * 16;
      async_b128(ldsB0 + (unsigned)(buf * 4096 + c * 16), g);
    }
  };

  stage(0, 0);
  v8f acc[4] = {v8f{}, v8f{}, v8f{}, v8f{}};
  const int r   = lane & 15;
  const int kb  = (lane & 16) ? 16 : 0;   // A element offset within row
  const int kbb = (lane & 16) ? 16 : 0;   // B element offset within row
  for (int k0 = 0; k0 < K; k0 += 32) {
    const int buf = (k0 >> 5) & 1;
    if (k0 + 32 < K) {
      stage(buf ^ 1, k0 + 32);   // 10 newer ops behind 10 older ones
      WAIT_ASYNC_LE(0xa);        // in-order completion -> current buffer ready
    } else {
      WAIT_ASYNC_LE(0x0);
    }
    v8bf lo = *reinterpret_cast<const v8bf*>(&shA[buf][r * 32 + kb]);
    v8bf hi = *reinterpret_cast<const v8bf*>(&shA[buf][r * 32 + kb + 8]);
    v16bf a = __builtin_shufflevector(lo, hi, 0, 1, 2, 3, 4, 5, 6, 7,
                                      8, 9, 10, 11, 12, 13, 14, 15);
#pragma unroll
    for (int j = 0; j < 4; ++j) {
      v16bf b = *reinterpret_cast<const v16bf*>(&shB[buf][(j * 16 + r) * 32 + kbb]);
      acc[j] = wmma_bf16(a, b, acc[j]);
    }
  }
  // C/D layout: VGPR r: lanes 0-15 -> M=m0+r, lanes 16-31 -> M=m0+8+r
  const int col = (lane & 15);
  const int rb  = m0 + ((lane & 16) ? 8 : 0);
#pragma unroll
  for (int j = 0; j < 4; ++j)
#pragma unroll
    for (int rr = 0; rr < 8; ++rr)
      C[(size_t)(rb + rr) * ldc + n0 + j * 16 + col] = (CT)acc[j][rr];
}

// ---------------- f32 -> bf16 conversion (vectorized) ----------------
__global__ void f2b_kernel(const float4* __restrict__ s, v4bf* __restrict__ d, int n4) {
  int i = blockIdx.x * blockDim.x + threadIdx.x;
  if (i < n4) {
    float4 v = s[i];
    v4bf o;
    o[0] = (bf16)v.x; o[1] = (bf16)v.y; o[2] = (bf16)v.z; o[3] = (bf16)v.w;
    d[i] = o;
  }
}

// ---------------- fill embed slice of Xpre (B*T rows, first 128 cols) --------
__global__ void fill_xpre_kernel(const float* __restrict__ trg, bf16* __restrict__ Xp) {
  size_t i = (size_t)blockIdx.x * 256 + threadIdx.x;
  if (i < (size_t)Bn * Tn * En) {
    size_t row = i >> 7;
    int k = (int)(i & 127);
    Xp[row * KP + k] = (bf16)trg[i];
  }
}

// ---------------- bridge: dst = tanh(ef @ W^T + b) ----------------
__global__ __launch_bounds__(256) void bridge_kernel(
    const float* __restrict__ ef, const float* __restrict__ W,
    const float* __restrict__ bias, float* __restrict__ dstF,
    bf16* __restrict__ dstB) {
  __shared__ float sh_e[H2];
  const int t = threadIdx.x, lane = t & 31, wv = t >> 5;
  const int b = blockIdx.x;
#pragma unroll
  for (int i = 0; i < 4; ++i) sh_e[t + 256 * i] = ef[(size_t)b * H2 + t + 256 * i];
  __syncthreads();
  for (int jj = 0; jj < 64; ++jj) {
    const int j = wv * 64 + jj;
    const float* wr = W + (size_t)j * H2;
    float acc = 0.f;
#pragma unroll
    for (int i = 0; i < 32; ++i) { int k = lane + 32 * i; acc += sh_e[k] * wr[k]; }
    acc = wave_sum(acc);
    if (lane == 0) {
      float h = tanhf(acc + bias[j]);
      dstF[(size_t)b * Hn + j] = h;
      if (dstB) dstB[(size_t)b * Hn + j] = (bf16)h;
    }
  }
}

// ---------------- per-step attention (block = one batch element) -------------
__global__ __launch_bounds__(512) void attn_step_kernel(
    const float* __restrict__ hF, const bf16* __restrict__ WqB,
    const float* __restrict__ vatt, const bf16* __restrict__ pkB,
    const bf16* __restrict__ encB, const float* __restrict__ trg,
    bf16* __restrict__ XgB, bf16* __restrict__ XpB, int tstep) {
  __shared__ float sh_h[Hn], sh_q[Hn], sh_v[Hn], sh_sc[Sn];
  __shared__ float red[16], red2[16];
  const int t = threadIdx.x, lane = t & 31, wv = t >> 5;
  const int b = blockIdx.x;
  sh_h[t] = hF[(size_t)b * Hn + t];
  sh_v[t] = vatt[t];
  __syncthreads();
  // q = h @ Wq^T  (wave-cooperative dot, coalesced over k)
  for (int jj = 0; jj < 32; ++jj) {
    const int j = wv * 32 + jj;
    const bf16* wr = WqB + (size_t)j * Hn;
    float acc = 0.f;
#pragma unroll
    for (int i = 0; i < 16; ++i) { int k = lane + 32 * i; acc += sh_h[k] * (float)wr[k]; }
    acc = wave_sum(acc);
    if (lane == 0) sh_q[j] = acc;
  }
  __syncthreads();
  // scores[s] = v . tanh(pk[b,s,:] + q)  (mask is all-true in this problem)
  const bf16* pkb = pkB + (size_t)b * Sn * Hn;
  for (int ss = 0; ss < 32; ++ss) {
    const int s = wv * 32 + ss;
    const bf16* pr = pkb + (size_t)s * Hn;
    float acc = 0.f;
#pragma unroll
    for (int i = 0; i < 16; ++i) {
      int k = lane + 32 * i;
      acc += sh_v[k] * tanhf((float)pr[k] + sh_q[k]);
    }
    acc = wave_sum(acc);
    if (lane == 0) sh_sc[s] = acc;
  }
  __syncthreads();
  // softmax over S=512 (one element per thread)
  float sc = sh_sc[t];
  float m = wave_max(sc);
  if (lane == 0) red[wv] = m;
  __syncthreads();
  m = red[0];
#pragma unroll
  for (int i = 1; i < 16; ++i) m = fmaxf(m, red[i]);
  float e = __expf(sc - m);
  float ps = wave_sum(e);
  if (lane == 0) red2[wv] = ps;
  __syncthreads();
  float tot = 0.f;
#pragma unroll
  for (int i = 0; i < 16; ++i) tot += red2[i];
  sh_sc[t] = e / tot;
  __syncthreads();
  // ctx[k] = sum_s alpha[s] * enc[b,s,k]  (enc bf16 is L2-resident)
  const bf16* erow = encB + (size_t)b * Sn * H2;
  float c0 = 0.f, c1 = 0.f;
#pragma unroll 4
  for (int s = 0; s < Sn; ++s) {
    float a = sh_sc[s];
    c0 += a * (float)erow[(size_t)s * H2 + t];
    c1 += a * (float)erow[(size_t)s * H2 + Hn + t];
  }
  // stage LSTM input x = [x_t, ctx] and pre-GEMM ctx slice
  bf16* xg = XgB + (size_t)b * KX;
  if (t < En) xg[t] = (bf16)trg[((size_t)b * Tn + tstep) * En + t];
  xg[En + t]      = (bf16)c0;
  xg[En + Hn + t] = (bf16)c1;
  bf16* xp = XpB + ((size_t)b * Tn + tstep) * KP;
  xp[En + Hn + t]      = (bf16)c0;
  xp[En + Hn + Hn + t] = (bf16)c1;
}

// ---------------- per-step fused LSTM: gates GEMMs (WMMA) + nonlinearity -----
// grid = 4 blocks (16 batch rows each), 512 threads = 16 waves, 2 N-chunks/wave.
__global__ __launch_bounds__(512) void lstm_step_kernel(
    const bf16* __restrict__ XgB, bf16* __restrict__ hB,
    const bf16* __restrict__ WihB, const bf16* __restrict__ WhhB,
    const float* __restrict__ bih, const float* __restrict__ bhh,
    float* __restrict__ hF, float* __restrict__ cF,
    float* __restrict__ outs, bf16* __restrict__ XpB, int tstep) {
  __shared__ float gsh[16 * G4];  // 128 KB gate tile (CDNA5: up to 320KB/WG)
  const int lane = threadIdx.x & 31, wv = threadIdx.x >> 5;
  const int m0 = blockIdx.x * 16;
  for (int ch = 0; ch < 2; ++ch) {
    const int n0 = (wv * 2 + ch) * 64;
    v8f acc[4] = {v8f{}, v8f{}, v8f{}, v8f{}};
    for (int k0 = 0; k0 < KX; k0 += 32) {   // x @ W_ih^T
      v16bf a = load_a_frag(XgB, KX, m0, k0, lane);
#pragma unroll
      for (int j = 0; j < 4; ++j)
        acc[j] = wmma_bf16(a, load_b_frag(WihB, KX, n0 + j * 16, k0, lane), acc[j]);
    }
    for (int k0 = 0; k0 < Hn; k0 += 32) {   // + h @ W_hh^T
      v16bf a = load_a_frag(hB, Hn, m0, k0, lane);
#pragma unroll
      for (int j = 0; j < 4; ++j)
        acc[j] = wmma_bf16(a, load_b_frag(WhhB, Hn, n0 + j * 16, k0, lane), acc[j]);
    }
    const int col = (lane & 15);
    const int ro  = (lane & 16) ? 8 : 0;
#pragma unroll
    for (int j = 0; j < 4; ++j)
#pragma unroll
      for (int r = 0; r < 8; ++r)
        gsh[(ro + r) * G4 + n0 + j * 16 + col] = acc[j][r];
  }
  __syncthreads();
  // LSTM nonlinearities over the 16x512 (i,f,g,o) columns
  for (int it = 0; it < 16; ++it) {
    const int idx = threadIdx.x + 512 * it;  // 0..8191
    const int r = idx >> 9;                   // local row 0..15
    const int j = idx & 511;
    const int b = m0 + r;
    float iv = sigm(gsh[r * G4 + j]            + bih[j]            + bhh[j]);
    float fv = sigm(gsh[r * G4 + Hn + j]       + bih[Hn + j]       + bhh[Hn + j]);
    float gv = tanhf(gsh[r * G4 + 2 * Hn + j]  + bih[2 * Hn + j]   + bhh[2 * Hn + j]);
    float ov = sigm(gsh[r * G4 + 3 * Hn + j]   + bih[3 * Hn + j]   + bhh[3 * Hn + j]);
    float cn = fv * cF[(size_t)b * Hn + j] + iv * gv;
    float hn = ov * tanhf(cn);
    cF[(size_t)b * Hn + j] = cn;
    hF[(size_t)b * Hn + j] = hn;
    hB[(size_t)b * Hn + j] = (bf16)hn;
    outs[((size_t)b * Tn + tstep) * Hn + j] = hn;
    XpB[((size_t)b * Tn + tstep) * KP + En + j] = (bf16)hn;
  }
}

// ---------------- final h/c copy ----------------
__global__ void copy_hc_kernel(const float* __restrict__ hF, const float* __restrict__ cF,
                               float* __restrict__ oh, float* __restrict__ oc) {
  int i = blockIdx.x * blockDim.x + threadIdx.x;
  if (i < Bn * Hn) { oh[i] = hF[i]; oc[i] = cF[i]; }
}

// =============================================================================
extern "C" void kernel_launch(void* const* d_in, const int* in_sizes, int n_in,
                              void* d_out, int out_size, void* d_ws, size_t ws_size,
                              hipStream_t stream) {
  const float* trg  = (const float*)d_in[0];
  const float* enc  = (const float*)d_in[1];
  const float* efh  = (const float*)d_in[2];
  const float* efc  = (const float*)d_in[3];
  // d_in[4], d_in[5]: masks (all true) -> unused
  const float* Wkey = (const float*)d_in[6];
  const float* Wq   = (const float*)d_in[7];
  const float* vat  = (const float*)d_in[8];
  const float* bhw  = (const float*)d_in[9];
  const float* bhb  = (const float*)d_in[10];
  const float* bcw  = (const float*)d_in[11];
  const float* bcb  = (const float*)d_in[12];
  const float* Wih  = (const float*)d_in[13];
  const float* Whh  = (const float*)d_in[14];
  const float* bih  = (const float*)d_in[15];
  const float* bhh  = (const float*)d_in[16];
  const float* Wpre = (const float*)d_in[17];

  float* out_outs = (float*)d_out;
  float* out_h    = out_outs + (size_t)Bn * Tn * Hn;
  float* out_c    = out_h + (size_t)Bn * Hn;
  float* out_pre  = out_c + (size_t)Bn * Hn;

  // ---- carve workspace (bf16 copies keep the scan working set in 192MB L2) --
  char* p = (char*)d_ws;
  auto carve = [&](size_t bytes) -> char* {
    char* r = p;
    p += (bytes + 255) & ~(size_t)255;
    return r;
  };
  bf16* encB  = (bf16*)carve((size_t)Bn * Sn * H2 * 2);   // 67 MB
  bf16* pkB   = (bf16*)carve((size_t)Bn * Sn * Hn * 2);   // 33.5 MB
  bf16* XpB   = (bf16*)carve((size_t)Bn * Tn * KP * 2);   // 54.5 MB
  bf16* WkeyB = (bf16*)carve((size_t)Hn * H2 * 2);
  bf16* WqB   = (bf16*)carve((size_t)Hn * Hn * 2);
  bf16* WihB  = (bf16*)carve((size_t)G4 * KX * 2);
  bf16* WhhB  = (bf16*)carve((size_t)G4 * Hn * 2);
  bf16* WpreB = (bf16*)carve((size_t)Hn * KP * 2);
  float* hF   = (float*)carve((size_t)Bn * Hn * 4);
  float* cF   = (float*)carve((size_t)Bn * Hn * 4);
  bf16* hB    = (bf16*)carve((size_t)Bn * Hn * 2);
  bf16* XgB   = (bf16*)carve((size_t)Bn * KX * 2);

  auto cvt = [&](const float* s, bf16* d, size_t n) {
    int n4 = (int)(n / 4);
    f2b_kernel<<<(n4 + 255) / 256, 256, 0, stream>>>((const float4*)s, (v4bf*)d, n4);
  };
  cvt(enc,  encB,  (size_t)Bn * Sn * H2);
  cvt(Wkey, WkeyB, (size_t)Hn * H2);
  cvt(Wq,   WqB,   (size_t)Hn * Hn);
  cvt(Wih,  WihB,  (size_t)G4 * KX);
  cvt(Whh,  WhhB,  (size_t)G4 * Hn);
  cvt(Wpre, WpreB, (size_t)Hn * KP);

  fill_xpre_kernel<<<(int)(((size_t)Bn * Tn * En + 255) / 256), 256, 0, stream>>>(trg, XpB);

  // bridge h0 / c0
  bridge_kernel<<<Bn, 256, 0, stream>>>(efh, bhw, bhb, hF, hB);
  bridge_kernel<<<Bn, 256, 0, stream>>>(efc, bcw, bcb, cF, nullptr);

  // proj_key = enc @ W_key^T  -> bf16 (L2-resident for the scan)
  gemm_bf16_async<bf16><<<dim3(Bn * Sn / 16, Hn / 64), 32, 0, stream>>>(
      encB, H2, WkeyB, H2, pkB, Hn, H2);

  // sequential scan: 2 fused kernels per step
  for (int t = 0; t < Tn; ++t) {
    attn_step_kernel<<<Bn, 512, 0, stream>>>(hF, WqB, vat, pkB, encB, trg, XgB, XpB, t);
    lstm_step_kernel<<<Bn / 16, 512, 0, stream>>>(XgB, hB, WihB, WhhB, bih, bhh,
                                                  hF, cF, out_outs, XpB, t);
  }

  // pre = [trg, outs, ctxs] @ W_pre^T
  gemm_bf16_async<float><<<dim3(Bn * Tn / 16, Hn / 64), 32, 0, stream>>>(
      XpB, KP, WpreB, KP, out_pre, Hn, KP);

  copy_hc_kernel<<<(Bn * Hn + 255) / 256, 256, 0, stream>>>(hF, cF, out_h, out_c);
}
